// Qwen3MoeForCausalLM_58102317580886
// MI455X (gfx1250) — compile-verified, matrix operand order checked by reading, not run on hardware
//
#include <hip/hip_runtime.h>
#include <hip/hip_bf16.h>

// ---- model constants (Qwen3-MoE toy config from the reference) ----
#define D_    2048
#define HD_   128
#define HQ_   16
#define HKV_  4
#define E_    8
#define F_    768
#define L_    2
#define V_    32000
#define S_    1024
#define EPS_  1e-6f
#define QKVN  ((HQ_ + 2 * HKV_) * HD_)   // 3072
#define ODIM  (HQ_ * HD_)                // 2048

typedef __bf16 bf16_t;
typedef bf16_t v16bf __attribute__((ext_vector_type(16)));
typedef bf16_t v8bf  __attribute__((ext_vector_type(8)));
typedef float  v8f   __attribute__((ext_vector_type(8)));

// ------------------------------------------------------------------
// reductions (wave32 + LDS across waves)
// ------------------------------------------------------------------
__device__ __forceinline__ float waveReduceSum(float v) {
  #pragma unroll
  for (int o = 16; o; o >>= 1) v += __shfl_xor(v, o, 32);
  return v;
}
__device__ __forceinline__ float waveReduceMax(float v) {
  #pragma unroll
  for (int o = 16; o; o >>= 1) v = fmaxf(v, __shfl_xor(v, o, 32));
  return v;
}
__device__ float blockReduceSum(float v) {
  __shared__ float s[32];
  const int lane = threadIdx.x & 31, w = threadIdx.x >> 5;
  v = waveReduceSum(v);
  if (lane == 0) s[w] = v;
  __syncthreads();
  const int nw = (blockDim.x + 31) >> 5;
  v = (threadIdx.x < (unsigned)nw) ? s[threadIdx.x] : 0.f;
  if (w == 0) v = waveReduceSum(v);
  if (threadIdx.x == 0) s[0] = v;
  __syncthreads();
  return s[0];
}
__device__ float blockReduceMax(float v) {
  __shared__ float s[32];
  const int lane = threadIdx.x & 31, w = threadIdx.x >> 5;
  v = waveReduceMax(v);
  if (lane == 0) s[w] = v;
  __syncthreads();
  const int nw = (blockDim.x + 31) >> 5;
  v = (threadIdx.x < (unsigned)nw) ? s[threadIdx.x] : -3.0e38f;
  if (w == 0) v = waveReduceMax(v);
  if (threadIdx.x == 0) s[0] = v;
  __syncthreads();
  return s[0];
}

__device__ __forceinline__ v8f wmma_bf16(v16bf a, v16bf b, v8f c) {
  // D(f32 16x16) = A(16x32 bf16) * B(32x16 bf16) + C
  return __builtin_amdgcn_wmma_f32_16x16x32_bf16(false, a, false, b,
                                                 (short)0, c, false, false);
}

// ------------------------------------------------------------------
// embedding gather
// ------------------------------------------------------------------
__global__ void q3m_embed(const float* __restrict__ embed,
                          const int* __restrict__ ids,
                          float* __restrict__ h) {
  const int i = blockIdx.x * 256 + threadIdx.x;       // i < S*D
  const int t = i >> 11;                              // D_ = 2^11
  const int d = i & (D_ - 1);
  h[i] = embed[(size_t)ids[t] * D_ + d];
}

// ------------------------------------------------------------------
// fused (a [+ b]) -> residual, RMSNorm -> x (f32) and x (bf16)
// ------------------------------------------------------------------
__global__ void __launch_bounds__(256)
q3m_add_rmsnorm(const float* __restrict__ a, const float* __restrict__ b,
                const float* __restrict__ w, float* __restrict__ resid_out,
                float* __restrict__ x_out, bf16_t* __restrict__ xbf_out) {
  const int t = blockIdx.x;
  float loc[8];
  float ss = 0.f;
  #pragma unroll
  for (int j = 0; j < 8; ++j) {
    const int d = threadIdx.x + j * 256;
    float r = a[(size_t)t * D_ + d];
    if (b) r += b[(size_t)t * D_ + d];
    loc[j] = r;
    ss += r * r;
    if (resid_out) resid_out[(size_t)t * D_ + d] = r;
  }
  ss = blockReduceSum(ss);
  const float inv = rsqrtf(ss * (1.0f / D_) + EPS_);
  #pragma unroll
  for (int j = 0; j < 8; ++j) {
    const int d = threadIdx.x + j * 256;
    const float xv = loc[j] * inv * w[d];
    if (x_out)   x_out[(size_t)t * D_ + d] = xv;
    if (xbf_out) xbf_out[(size_t)t * D_ + d] = (bf16_t)xv;
  }
}

// ------------------------------------------------------------------
// LDS-tiled GEMM: C[M,N] f32 = A[M,K] bf16 @ B[K,N] f32
// grid (N/128, M/128), 256 threads = 8 waves (4 row-groups x 2 col-groups)
// per K-step(32):
//   A 128x32 bf16 staged via GLOBAL_LOAD_ASYNC_TO_LDS_B128 (ASYNCcnt)
//   B 32x128 f32 -> v_cvt_pk_bf16_f32 -> transposed [n][k] in LDS
//   each wave runs 8 x v_wmma_f32_16x16x32_bf16 from ds_load_b128 frags
// LDS row stride = 40 bf16 (80B, multiple of 16B, bank-conflict padding).
// ------------------------------------------------------------------
#define TSTR 40
__global__ void __launch_bounds__(256)
q3m_gemm_bf16(const bf16_t* __restrict__ A, const float* __restrict__ B,
              float* __restrict__ C, int M, int N, int K) {
  __shared__ alignas(16) bf16_t As[128 * TSTR];   // [row][k]
  __shared__ alignas(16) bf16_t Bs[128 * TSTR];   // [col][k] (transposed)

  const int tid  = threadIdx.x;
  const int wv   = tid >> 5;
  const int lane = tid & 31;
  const int half = lane >> 4;
  const int rc   = lane & 15;
  const int wr   = wv >> 1;            // 0..3 : 32-row group
  const int wc   = wv & 1;             // 0..1 : 64-col group
  const int m0   = blockIdx.y * 128;
  const int n0   = blockIdx.x * 128;

  // staging assignments
  const int arow  = tid >> 1;          // 0..127
  const int akoff = (tid & 1) * 16;    // 0 / 16
  const int bkg   = tid >> 4;          // 0..15 -> k pair bkg*2
  const int bng   = tid & 15;          // 0..15 -> cols bng*8..+8

  v8f acc[2][4];
  #pragma unroll
  for (int mt = 0; mt < 2; ++mt)
    #pragma unroll
    for (int nt = 0; nt < 4; ++nt) acc[mt][nt] = (v8f){};

  const bf16_t* aglob = A + (size_t)(m0 + arow) * K + akoff;
  const float*  bglob = B + (size_t)(bkg * 2) * N + n0 + bng * 8;
  // LDS byte address of this thread's A staging slot (flat addr low 32 bits
  // are the LDS offset per the ISA aperture rule)
  const unsigned lds_a = (unsigned)(size_t)(As + arow * TSTR + akoff);

  for (int k0 = 0; k0 < K; k0 += 32) {
    // ---- B: global -> registers (with prefetch of next panel) ----
    const float* b0 = bglob + (size_t)k0 * N;
    const float* b1 = b0 + N;
    float4 br0a = ((const float4*)b0)[0], br0b = ((const float4*)b0)[1];
    float4 br1a = ((const float4*)b1)[0], br1b = ((const float4*)b1)[1];
    if (k0 + 32 < K) {
      __builtin_prefetch(aglob + k0 + 32, 0, 1);
      __builtin_prefetch(b0 + (size_t)32 * N, 0, 1);
    }
    __syncthreads();   // previous compute done before overwriting LDS
    // ---- A: async copy global -> LDS (bf16 passthrough, no VGPR trip) ----
    {
      const bf16_t* ag = aglob + k0;
      asm volatile(
          "global_load_async_to_lds_b128 %0, %1, off\n\t"
          "global_load_async_to_lds_b128 %2, %3, off"
          :
          : "v"(lds_a), "v"(ag), "v"(lds_a + 16u), "v"(ag + 8)
          : "memory");
    }
    // ---- B: registers -> LDS transposed (packed bf16 pairs) ----
    {
      const float r0[8] = {br0a.x, br0a.y, br0a.z, br0a.w,
                           br0b.x, br0b.y, br0b.z, br0b.w};
      const float r1[8] = {br1a.x, br1a.y, br1a.z, br1a.w,
                           br1b.x, br1b.y, br1b.z, br1b.w};
      #pragma unroll
      for (int j = 0; j < 8; ++j) {
        union { unsigned u; bf16_t h[2]; } pk;
        pk.h[0] = (bf16_t)r0[j];          // k = bkg*2
        pk.h[1] = (bf16_t)r1[j];          // k = bkg*2+1
        *(unsigned*)(Bs + (bng * 8 + j) * TSTR + bkg * 2) = pk.u;
      }
    }
    asm volatile("s_wait_asynccnt 0x0" ::: "memory");  // A tile landed
    __syncthreads();   // staging visible to all waves
    // ---- LDS fragments + 8 WMMA ----
    union F { v16bf v; v8bf h[2]; };
    F afr[2], bfr[4];
    #pragma unroll
    for (int mt = 0; mt < 2; ++mt) {
      const bf16_t* ap = As + (wr * 32 + mt * 16 + rc) * TSTR;
      afr[mt].h[0] = *(const v8bf*)(ap + half * 8);
      afr[mt].h[1] = *(const v8bf*)(ap + 16 + half * 8);
    }
    #pragma unroll
    for (int nt = 0; nt < 4; ++nt) {
      const bf16_t* bp = Bs + (wc * 64 + nt * 16 + rc) * TSTR;
      bfr[nt].h[0] = *(const v8bf*)(bp + half * 8);
      bfr[nt].h[1] = *(const v8bf*)(bp + 16 + half * 8);
    }
    #pragma unroll
    for (int mt = 0; mt < 2; ++mt)
      #pragma unroll
      for (int nt = 0; nt < 4; ++nt)
        acc[mt][nt] = wmma_bf16(afr[mt].v, bfr[nt].v, acc[mt][nt]);
  }

  // ---- store C ----
  #pragma unroll
  for (int mt = 0; mt < 2; ++mt)
    #pragma unroll
    for (int nt = 0; nt < 4; ++nt) {
      const int col = n0 + wc * 64 + nt * 16 + rc;
      #pragma unroll
      for (int i = 0; i < 8; ++i) {
        const int row = m0 + wr * 32 + mt * 16 + i + 8 * half;
        C[(size_t)row * N + col] = acc[mt][nt][i];
      }
    }
}

// ------------------------------------------------------------------
// per-head q/k RMSNorm + RoPE (theta = 1e6), writes bf16 q/k and
// V transposed per kv-head: vt[hkv][d][t]
// ------------------------------------------------------------------
__global__ void __launch_bounds__(128)
q3m_qkv_rope(const float* __restrict__ qkv, const float* __restrict__ qnw,
             const float* __restrict__ knw, const int* __restrict__ pos_ids,
             bf16_t* __restrict__ qb, bf16_t* __restrict__ kb,
             bf16_t* __restrict__ vt) {
  const int t = blockIdx.x, head = blockIdx.y, d = threadIdx.x;
  float v = qkv[(size_t)t * QKVN + head * HD_ + d];
  const bool isq = head < HQ_;
  const bool isk = !isq && head < HQ_ + HKV_;
  const float ss = blockReduceSum(v * v);
  __shared__ float tile[HD_];
  if (isq || isk) {
    v = v * rsqrtf(ss * (1.0f / HD_) + EPS_) * (isq ? qnw[d] : knw[d]);
    tile[d] = v;
  }
  __syncthreads();
  if (isq || isk) {
    const float pos = (float)pos_ids[t];
    const int fi = d & 63;
    const float freq = __expf(-13.815510557964274f * (float)fi * (1.0f / 64.0f));
    const float ang = pos * freq;
    const float other = (d < 64) ? -tile[d + 64] : tile[d - 64];
    v = v * __cosf(ang) + other * __sinf(ang);
  }
  if (isq) {
    qb[(size_t)t * (HQ_ * HD_) + head * HD_ + d] = (bf16_t)v;
  } else if (isk) {
    kb[(size_t)t * (HKV_ * HD_) + (head - HQ_) * HD_ + d] = (bf16_t)v;
  } else {
    const int hv = head - HQ_ - HKV_;
    vt[((size_t)hv * HD_ + d) * S_ + t] = (bf16_t)v;   // transposed store
  }
}

// ------------------------------------------------------------------
// scores[h, q, k] = (q . k) / sqrt(HD); only lower-triangular tiles
// ------------------------------------------------------------------
__global__ void __launch_bounds__(32)
q3m_attn_scores(const bf16_t* __restrict__ qb, const bf16_t* __restrict__ kb,
                float* __restrict__ scores) {
  const int kt = blockIdx.x, qt = blockIdx.y, head = blockIdx.z;
  if (kt > qt) return;
  const int lane = threadIdx.x;
  const int half = lane >> 4, rc = lane & 15;
  const int hkv = head >> 2;
  const bf16_t* qrow = qb + (size_t)(qt * 16 + rc) * (HQ_ * HD_)  + head * HD_;
  const bf16_t* krow = kb + (size_t)(kt * 16 + rc) * (HKV_ * HD_) + hkv * HD_;
  v8f acc = {};
  #pragma unroll
  for (int k0 = 0; k0 < HD_; k0 += 32) {
    union { v16bf v; v8bf h[2]; } a, b;
    a.h[0] = *(const v8bf*)(qrow + k0 + half * 8);
    a.h[1] = *(const v8bf*)(qrow + k0 + 16 + half * 8);
    b.h[0] = *(const v8bf*)(krow + k0 + half * 8);
    b.h[1] = *(const v8bf*)(krow + k0 + 16 + half * 8);
    acc = wmma_bf16(a.v, b.v, acc);
  }
  const float sc = 0.08838834764831845f;   // 1/sqrt(128)
  float* srow = scores + (size_t)head * S_ * S_;
  #pragma unroll
  for (int i = 0; i < 8; ++i) {
    const int r = qt * 16 + i + 8 * half;
    const int c = kt * 16 + rc;
    srow[(size_t)r * S_ + c] = acc[i] * sc;
  }
}

// ------------------------------------------------------------------
// causal row softmax -> bf16 probs (zero-padded to 32-wide K boundary)
// ------------------------------------------------------------------
__global__ void __launch_bounds__(256)
q3m_softmax_causal(const float* __restrict__ scores, bf16_t* __restrict__ pbf) {
  const int row = blockIdx.x, head = blockIdx.y;
  const float* srow = scores + (size_t)head * S_ * S_ + (size_t)row * S_;
  bf16_t* prow = pbf + (size_t)head * S_ * S_ + (size_t)row * S_;
  const int n = row + 1;
  float mx = -1e30f;
  for (int c = threadIdx.x; c < n; c += 256) mx = fmaxf(mx, srow[c]);
  mx = blockReduceMax(mx);
  float sum = 0.f;
  for (int c = threadIdx.x; c < n; c += 256) sum += __expf(srow[c] - mx);
  sum = blockReduceSum(sum);
  const float inv = 1.f / sum;
  for (int c = threadIdx.x; c < n; c += 256)
    prow[c] = (bf16_t)(__expf(srow[c] - mx) * inv);
  const int qt = row >> 4;
  const int kpad = (((qt + 1) * 16) + 31) & ~31;
  for (int c = n + threadIdx.x; c < kpad; c += 256) prow[c] = (bf16_t)0.f;
}

// ------------------------------------------------------------------
// O[q, head, d] = P[q, :] @ V[:, d] per head; V is pre-transposed so
// both fragments are contiguous 16B loads
// ------------------------------------------------------------------
__global__ void __launch_bounds__(32)
q3m_attn_av(const bf16_t* __restrict__ pbf, const bf16_t* __restrict__ vt,
            float* __restrict__ out) {
  const int dt = blockIdx.x, qt = blockIdx.y, head = blockIdx.z;
  const int lane = threadIdx.x, half = lane >> 4, rc = lane & 15;
  const int hkv = head >> 2;
  const bf16_t* prow = pbf + (size_t)head * S_ * S_ + (size_t)(qt * 16 + rc) * S_;
  const bf16_t* vrow = vt + ((size_t)hkv * HD_ + dt * 16 + rc) * S_;
  const int kpad = (((qt + 1) * 16) + 31) & ~31;
  v8f acc = {};
  for (int k0 = 0; k0 < kpad; k0 += 32) {
    union { v16bf v; v8bf h[2]; } a, b;
    a.h[0] = *(const v8bf*)(prow + k0 + half * 8);
    a.h[1] = *(const v8bf*)(prow + k0 + 16 + half * 8);
    b.h[0] = *(const v8bf*)(vrow + k0 + half * 8);
    b.h[1] = *(const v8bf*)(vrow + k0 + 16 + half * 8);
    acc = wmma_bf16(a.v, b.v, acc);
  }
  #pragma unroll
  for (int i = 0; i < 8; ++i)
    out[(size_t)(qt * 16 + i + 8 * half) * (HQ_ * HD_) + head * HD_ + dt * 16 + rc]
        = acc[i];
}

// ------------------------------------------------------------------
// f32 -> bf16 elementwise
// ------------------------------------------------------------------
__global__ void q3m_f2bf(const float* __restrict__ in, bf16_t* __restrict__ out,
                         int n) {
  const int i = blockIdx.x * 256 + threadIdx.x;
  if (i < n) out[i] = (bf16_t)in[i];
}

// ------------------------------------------------------------------
// router: softmax over 8 experts, top-2, renormalize -> combine[T,E]
// ------------------------------------------------------------------
__global__ void __launch_bounds__(256)
q3m_gate(const float* __restrict__ x, const float* __restrict__ gw,
         float* __restrict__ combine) {
  const int t = blockIdx.x;
  const int w = threadIdx.x >> 5, lane = threadIdx.x & 31;
  float s = 0.f;
  for (int d = lane; d < D_; d += 32)
    s += x[(size_t)t * D_ + d] * gw[(size_t)d * E_ + w];
  s = waveReduceSum(s);
  __shared__ float lg[E_];
  if (lane == 0) lg[w] = s;
  __syncthreads();
  if (threadIdx.x == 0) {
    float mx = lg[0];
    #pragma unroll
    for (int e = 1; e < E_; ++e) mx = fmaxf(mx, lg[e]);
    float p[E_];
    float sum = 0.f;
    #pragma unroll
    for (int e = 0; e < E_; ++e) { p[e] = __expf(lg[e] - mx); sum += p[e]; }
    #pragma unroll
    for (int e = 0; e < E_; ++e) p[e] /= sum;
    int i1 = 0;
    #pragma unroll
    for (int e = 1; e < E_; ++e) if (p[e] > p[i1]) i1 = e;
    int i2 = (i1 == 0) ? 1 : 0;
    #pragma unroll
    for (int e = 0; e < E_; ++e) if (e != i1 && p[e] > p[i2]) i2 = e;
    const float rn = p[i1] + p[i2];
    #pragma unroll
    for (int e = 0; e < E_; ++e)
      combine[(size_t)t * E_ + e] =
          (e == i1) ? p[i1] / rn : (e == i2) ? p[i2] / rn : 0.f;
  }
}

// ------------------------------------------------------------------
// SiLU(gate) * up -> bf16 activations
// ------------------------------------------------------------------
__global__ void q3m_silu_mul(const float* __restrict__ gu,
                             bf16_t* __restrict__ act) {
  const int i = blockIdx.x * 256 + threadIdx.x;   // i < S*F
  const int t = i / F_, f = i - t * F_;
  const float g = gu[(size_t)t * 2 * F_ + f];
  const float u = gu[(size_t)t * 2 * F_ + F_ + f];
  const float sg = g / (1.f + __expf(-g));
  act[i] = (bf16_t)(sg * u);
}

// ------------------------------------------------------------------
// h += combine[t, e] * down_e
// ------------------------------------------------------------------
__global__ void q3m_moe_accum(const float* __restrict__ dtmp,
                              const float* __restrict__ combine, int e,
                              float* __restrict__ h) {
  const int i = blockIdx.x * 256 + threadIdx.x;
  const int t = i >> 11;
  h[i] += combine[(size_t)t * E_ + e] * dtmp[i];
}

// ------------------------------------------------------------------
// driver
// ------------------------------------------------------------------
extern "C" void kernel_launch(void* const* d_in, const int* in_sizes, int n_in,
                              void* d_out, int out_size, void* d_ws,
                              size_t ws_size, hipStream_t stream) {
  (void)in_sizes; (void)n_in; (void)out_size; (void)ws_size;
  const float* embed  = (const float*)d_in[0];
  const float* ln1_w  = (const float*)d_in[1];
  const float* qkv_w  = (const float*)d_in[2];
  const float* q_norm = (const float*)d_in[3];
  const float* k_norm = (const float*)d_in[4];
  const float* o_w    = (const float*)d_in[5];
  const float* ln2_w  = (const float*)d_in[6];
  const float* gate_w = (const float*)d_in[7];
  const float* w_gu   = (const float*)d_in[8];
  const float* w_dn   = (const float*)d_in[9];
  const float* fn_w   = (const float*)d_in[10];
  const int*   ids    = (const int*)d_in[11];
  const int*   pos    = (const int*)d_in[12];

  char* p = (char*)d_ws;
  auto alloc = [&](size_t bytes) {
    char* r = p;
    p += (bytes + 255) & ~(size_t)255;
    return r;
  };
  float*  hbuf   = (float*)alloc((size_t)S_ * D_ * 4);
  float*  resid  = (float*)alloc((size_t)S_ * D_ * 4);
  float*  xbuf   = (float*)alloc((size_t)S_ * D_ * 4);
  bf16_t* xbf    = (bf16_t*)alloc((size_t)S_ * D_ * 2);
  float*  qkvb   = (float*)alloc((size_t)S_ * QKVN * 4);
  bf16_t* qb     = (bf16_t*)alloc((size_t)S_ * HQ_ * HD_ * 2);
  bf16_t* kb     = (bf16_t*)alloc((size_t)S_ * HKV_ * HD_ * 2);
  bf16_t* vt     = (bf16_t*)alloc((size_t)HKV_ * HD_ * S_ * 2);
  float*  scores = (float*)alloc((size_t)HQ_ * S_ * S_ * 4);
  bf16_t* pbf    = (bf16_t*)alloc((size_t)HQ_ * S_ * S_ * 2);
  float*  aout   = (float*)alloc((size_t)S_ * HQ_ * HD_ * 4);
  bf16_t* aobf   = (bf16_t*)alloc((size_t)S_ * HQ_ * HD_ * 2);
  float*  oout   = (float*)alloc((size_t)S_ * D_ * 4);
  float*  comb   = (float*)alloc((size_t)S_ * E_ * 4);
  float*  gu     = (float*)alloc((size_t)S_ * 2 * F_ * 4);
  bf16_t* actbf  = (bf16_t*)alloc((size_t)S_ * F_ * 2);
  float*  dtmp   = (float*)alloc((size_t)S_ * D_ * 4);

  q3m_embed<<<(S_ * D_) / 256, 256, 0, stream>>>(embed, ids, hbuf);

  for (int l = 0; l < L_; ++l) {
    q3m_add_rmsnorm<<<S_, 256, 0, stream>>>(hbuf, l ? resid : nullptr,
                                            ln1_w + (size_t)l * D_, resid, xbuf,
                                            xbf);
    q3m_gemm_bf16<<<dim3(QKVN / 128, S_ / 128), 256, 0, stream>>>(
        xbf, qkv_w + (size_t)l * D_ * QKVN, qkvb, S_, QKVN, D_);
    q3m_qkv_rope<<<dim3(S_, HQ_ + 2 * HKV_), HD_, 0, stream>>>(
        qkvb, q_norm + (size_t)l * HD_, k_norm + (size_t)l * HD_, pos, qb, kb,
        vt);
    q3m_attn_scores<<<dim3(S_ / 16, S_ / 16, HQ_), 32, 0, stream>>>(qb, kb,
                                                                    scores);
    q3m_softmax_causal<<<dim3(S_, HQ_), 256, 0, stream>>>(scores, pbf);
    q3m_attn_av<<<dim3(HD_ / 16, S_ / 16, HQ_), 32, 0, stream>>>(pbf, vt,
                                                                 aout);
    q3m_f2bf<<<(S_ * ODIM) / 256, 256, 0, stream>>>(aout, aobf, S_ * ODIM);
    q3m_gemm_bf16<<<dim3(D_ / 128, S_ / 128), 256, 0, stream>>>(
        aobf, o_w + (size_t)l * ODIM * D_, oout, S_, D_, ODIM);
    q3m_add_rmsnorm<<<S_, 256, 0, stream>>>(oout, resid,
                                            ln2_w + (size_t)l * D_, resid, xbuf,
                                            xbf);
    q3m_gate<<<S_, 256, 0, stream>>>(xbuf, gate_w + (size_t)l * D_ * E_, comb);
    hipMemsetAsync(hbuf, 0, (size_t)S_ * D_ * 4, stream);
    for (int e = 0; e < E_; ++e) {
      q3m_gemm_bf16<<<dim3((2 * F_) / 128, S_ / 128), 256, 0, stream>>>(
          xbf, w_gu + ((size_t)l * E_ + e) * D_ * 2 * F_, gu, S_, 2 * F_, D_);
      q3m_silu_mul<<<(S_ * F_) / 256, 256, 0, stream>>>(gu, actbf);
      q3m_gemm_bf16<<<dim3(D_ / 128, S_ / 128), 256, 0, stream>>>(
          actbf, w_dn + ((size_t)l * E_ + e) * F_ * D_, dtmp, S_, D_, F_);
      q3m_moe_accum<<<(S_ * D_) / 256, 256, 0, stream>>>(dtmp, comb, e, hbuf);
    }
  }
  q3m_add_rmsnorm<<<S_, 256, 0, stream>>>(hbuf, resid, fn_w, nullptr,
                                          (float*)d_out, nullptr);
}
